// ForwardSim_79121887527499
// MI455X (gfx1250) — matrix operand order repaired and checked by reading, not compiled, not run on hardware
//
#include <hip/hip_runtime.h>

typedef __attribute__((ext_vector_type(16))) __bf16 v16bf;
typedef __attribute__((ext_vector_type(8)))  __bf16 v8bf;
typedef __attribute__((ext_vector_type(8)))  float  v8f;

#define B_TOT    32768
#define T_STEPS  50
#define NFEAT    14
#define NMD      3
#define NROWS    64      // batch rows per workgroup
#define NTHREADS 256     // 8 waves of 32
#define H1_STRIDE 136    // 128 + 8 pad (bf16 elems); 272B row: 16B aligned, conflict-free

#if defined(__has_builtin)
#  if __has_builtin(__builtin_amdgcn_global_load_async_to_lds_b32)
#    define HAVE_ASYNC 1
#  endif
#endif
#ifndef HAVE_ASYNC
#  define HAVE_ASYNC 0
#endif

#define AS_GLOBAL __attribute__((address_space(1)))
#define AS_LDS    __attribute__((address_space(3)))

__device__ __forceinline__ float leaky(float x) { return x >= 0.f ? x : 0.3f * x; }

#if HAVE_ASYNC
__device__ __forceinline__ void async_gather_b32(void* lds, const void* g) {
  __builtin_amdgcn_global_load_async_to_lds_b32(
      (AS_GLOBAL int*)(void*)g, (AS_LDS int*)(void*)lds, 0, 0);
}
__device__ __forceinline__ void wait_async_le9() {
#  if __has_builtin(__builtin_amdgcn_s_wait_asynccnt)
  __builtin_amdgcn_s_wait_asynccnt(9);
#  else
  asm volatile("s_wait_asynccnt 9" ::: "memory");
#  endif
}
__device__ __forceinline__ void wait_async_le0() {
#  if __has_builtin(__builtin_amdgcn_s_wait_asynccnt)
  __builtin_amdgcn_s_wait_asynccnt(0);
#  else
  asm volatile("s_wait_asynccnt 0" ::: "memory");
#  endif
}
// gather the 9 per-row state words for one timestep into LDS via async DMA
__device__ __forceinline__ void issue_tile(float* dst9, const float* ip, const float* mp) {
  async_gather_b32(dst9 + 0, ip + 1);    // f_veh_v
  async_gather_b32(dst9 + 1, ip + 2);    // m_veh_v
  async_gather_b32(dst9 + 2, ip + 4);    // f_glob_x
  async_gather_b32(dst9 + 3, ip + 5);    // m_glob_x
  async_gather_b32(dst9 + 4, ip + 12);   // f_veh_a
  async_gather_b32(dst9 + 5, ip + 13);   // m_exists
  async_gather_b32(dst9 + 6, mp + 0);
  async_gather_b32(dst9 + 7, mp + 1);
  async_gather_b32(dst9 + 8, mp + 2);
}
#endif

// A fragment (16x32 bf16) from LDS, row-major with H1_STRIDE.
// lane<16: row m=lane, K = kf*32 + {0..7,16..23}; lane>=16: row m=lane-16, K = kf*32 + {8..15,24..31}
__device__ __forceinline__ v16bf load_a_frag(const __bf16* base, int lane, int mt, int kf) {
  int m   = mt * 16 + (lane & 15);
  int klo = kf * 32 + ((lane < 16) ? 0 : 8);
  const __bf16* p = base + m * H1_STRIDE + klo;
  v8bf lo = *(const v8bf*)(p);        // ds_load_b128
  v8bf hi = *(const v8bf*)(p + 16);   // ds_load_b128
  return __builtin_shufflevector(lo, hi, 0,1,2,3,4,5,6,7,8,9,10,11,12,13,14,15);
}

// B fragment (32x16 bf16): lane holds column n; lane<16: K=kf*32+{0..7,16..23}, lane>=16: K=kf*32+{8..15,24..31}
__device__ __forceinline__ v16bf load_b_frag(const float* __restrict__ W, int n, int kf, int lane) {
  int kbase = kf * 32 + ((lane < 16) ? 0 : 8);
  v16bf b;
#pragma unroll
  for (int i = 0; i < 16; ++i) b[i] = (__bf16)W[(size_t)(kbase + i + ((i < 8) ? 0 : 8)) * 128 + n];
  return b;
}

__device__ __forceinline__ v8f wmma_bf16(v16bf a, v16bf b, v8f c) {
  return __builtin_amdgcn_wmma_f32_16x16x32_bf16(false, a, false, b, (short)0, c, false, false);
}

__global__ __launch_bounds__(NTHREADS)
void forward_sim_kernel(const float* __restrict__ proj, const float* __restrict__ enc,
                        const float* __restrict__ idm,  const float* __restrict__ mcg,
                        const float* __restrict__ env_mean, const float* __restrict__ env_var,
                        const float* __restrict__ W1, const float* __restrict__ b1,
                        const float* __restrict__ W2, const float* __restrict__ b2,
                        const float* __restrict__ W3, const float* __restrict__ b3,
                        float* __restrict__ outDisp, float* __restrict__ outAct)
{
  __shared__ float s_pre[NROWS][128];                         // 32768 B
  __shared__ float s_w1env[11][128];                          //  5632 B
  __shared__ __align__(16) __bf16 s_h1[NROWS][H1_STRIDE];     // 17408 B (also precompute staging)
  __shared__ float s_env[NROWS][12];                          //  3072 B (env[0..7] + mc[8..10])
  __shared__ float s_acc3[NROWS];                             //   256 B
  __shared__ float s_idm[2][NROWS][9];                        //  4608 B (async double buffer)

  const int tid  = threadIdx.x;
  const int lane = tid & 31;
  const int wid  = tid >> 5;
  const int b0   = blockIdx.x * NROWS;
  const int n    = wid * 16 + (lane & 15);   // output neuron column owned by this lane

  // stage W1 env/mc rows (11 x 128)
  for (int i = tid; i < 11 * 128; i += NTHREADS)
    ((float*)s_w1env)[i] = W1[(size_t)(256 + (i >> 7)) * 128 + (i & 127)];

  // per-row state (rows owned by threads 0..63)
  float ego_v = 0.f, ego_a = 0.f, ego_x = 0.f, disp = 0.f;
  float emean[8], istd[8];
  if (tid < NROWS) {
#pragma unroll
    for (int j = 0; j < 8; ++j) { emean[j] = env_mean[j]; istd[j] = rsqrtf(env_var[j]); }
    const int b = b0 + tid;
    const float* ip0 = idm + (size_t)b * T_STEPS * NFEAT;
    ego_v = ip0[0]; ego_a = ip0[11]; ego_x = ip0[3];
    outDisp[(size_t)b * (T_STEPS + 1)] = 0.f;     // displacement_seq[:,0] = 0
#if HAVE_ASYNC
    // prologue: prefetch steps 0 and 1 (hidden behind the precompute below)
    issue_tile(&s_idm[0][tid][0], idm + (size_t)b * T_STEPS * NFEAT,            mcg + (size_t)b * T_STEPS * NMD);
    issue_tile(&s_idm[1][tid][0], idm + ((size_t)b * T_STEPS + 1) * NFEAT,      mcg + ((size_t)b * T_STEPS + 1) * NMD);
#endif
  }

  // ---------- precompute pre = [proj, enc] @ W1[0:256] + b1 via WMMA ----------
  const v8f vzero = {0.f, 0.f, 0.f, 0.f, 0.f, 0.f, 0.f, 0.f};
  v8f cpre[4];
#pragma unroll
  for (int mt = 0; mt < 4; ++mt) cpre[mt] = vzero;

  for (int pass = 0; pass < 2; ++pass) {
    const float* src = (pass == 0) ? proj : enc;
    __syncthreads();                         // protect staging buffer reuse
    for (int i = tid; i < NROWS * 128; i += NTHREADS) {
      int m = i >> 7, k = i & 127;
      s_h1[m][k] = (__bf16)src[(size_t)(b0 + m) * 128 + k];
    }
    __syncthreads();
    const float* Wp = W1 + (size_t)pass * 128 * 128;
#pragma unroll
    for (int kf = 0; kf < 4; ++kf) {
      v16bf bf = load_b_frag(Wp, n, kf, lane);
#pragma unroll
      for (int mt = 0; mt < 4; ++mt) {
        v16bf af = load_a_frag(&s_h1[0][0], lane, mt, kf);
        cpre[mt] = wmma_bf16(af, bf, cpre[mt]);
      }
    }
  }
  {
    float b1n = b1[n];
#pragma unroll
    for (int mt = 0; mt < 4; ++mt)
#pragma unroll
      for (int v = 0; v < 8; ++v)
        s_pre[mt * 16 + v + ((lane < 16) ? 0 : 8)][n] = cpre[mt][v] + b1n;
  }

  // persistent W2 B-fragments (held in VGPRs across all 50 steps)
  v16bf w2f[4];
#pragma unroll
  for (int kf = 0; kf < 4; ++kf) w2f[kf] = load_b_frag(W2, n, kf, lane);

  const float b2n = b2[n];
  const float w3n = W3[n];
  const float b3v = b3[0];

  // ------------------------------ rollout (3 barriers/step) ------------------------------
  for (int t = 0; t < T_STEPS; ++t) {
    const int buf = t & 1;
    if (tid < NROWS) {
      s_acc3[tid] = 0.f;
      const int b = b0 + tid;
      float fv, mv, fx, mx, fa, me, c0, c1, c2;
#if HAVE_ASYNC
      if (t < T_STEPS - 1) wait_async_le9();   // batch(t) done; batch(t+1) may be in flight
      else                 wait_async_le0();
      const float* rp = &s_idm[buf][tid][0];
      fv = rp[0]; mv = rp[1]; fx = rp[2]; mx = rp[3]; fa = rp[4]; me = rp[5];
      c0 = rp[6]; c1 = rp[7]; c2 = rp[8];
#else
      const float* ip = idm + ((size_t)b * T_STEPS + t) * NFEAT;
      const float* mp = mcg + ((size_t)b * T_STEPS + t) * NMD;
      fv = ip[1]; mv = ip[2]; fx = ip[4]; mx = ip[5]; fa = ip[12]; me = ip[13];
      c0 = mp[0]; c1 = mp[1]; c2 = mp[2];
#endif
      float e[8];
      e[0] = ego_v;  e[1] = fv;  e[2] = ego_a;  e[3] = fa;
      e[4] = ego_v - fv;
      e[5] = fx - ego_x;
      e[6] = (ego_v - mv) * me;                               // DUMMY_DV = 0
      e[7] = (mx - ego_x) * me + (1.f - me) * 50.f;           // DUMMY_DX = 50
#pragma unroll
      for (int j = 0; j < 8; ++j) s_env[tid][j] = (e[j] - emean[j]) * istd[j];
      s_env[tid][8] = c0; s_env[tid][9] = c1; s_env[tid][10] = c2;
#if HAVE_ASYNC
      const int t2 = t + 2;                    // refill the buffer we just consumed
      if (t2 < T_STEPS)
        issue_tile(&s_idm[buf][tid][0],
                   idm + ((size_t)b * T_STEPS + t2) * NFEAT,
                   mcg + ((size_t)b * T_STEPS + t2) * NMD);
#endif
    }
    __syncthreads();   // (A) env ready

    // h1 = leaky(pre + env*W1env + mc*W1mc)  -> bf16 LDS
    {
      int k = tid & 127;
      int mbase = (tid >> 7) * 32;
      float wv[11];
#pragma unroll
      for (int j = 0; j < 11; ++j) wv[j] = s_w1env[j][k];
      for (int mm = 0; mm < 32; ++mm) {
        int m = mbase + mm;
        float s = s_pre[m][k];
#pragma unroll
        for (int j = 0; j < 8; ++j) s += s_env[m][j] * wv[j];
        s += s_env[m][8] * wv[8] + s_env[m][9] * wv[9] + s_env[m][10] * wv[10];
        s_h1[m][k] = (__bf16)leaky(s);
      }
    }
    __syncthreads();   // (B) h1 ready

    // layer 2 via WMMA (16 per wave per step) + fused layer-3 reduction
#pragma unroll
    for (int mt = 0; mt < 4; ++mt) {
      v8f c = vzero;
#pragma unroll
      for (int kf = 0; kf < 4; ++kf) {
        v16bf af = load_a_frag(&s_h1[0][0], lane, mt, kf);
        c = wmma_bf16(af, w2f[kf], c);
      }
#pragma unroll
      for (int v = 0; v < 8; ++v) {
        float h   = leaky(c[v] + b2n);
        float val = h * w3n;
        val += __shfl_xor(val, 1, 32);
        val += __shfl_xor(val, 2, 32);
        val += __shfl_xor(val, 4, 32);
        val += __shfl_xor(val, 8, 32);       // sum the 16 neurons this half-wave holds
        if ((lane & 15) == 0)
          atomicAdd(&s_acc3[mt * 16 + v + ((lane < 16) ? 0 : 8)], val);
      }
    }
    __syncthreads();   // (C) acc3 ready

    if (tid < NROWS) {
      const int b = b0 + tid;
      float a_new = s_acc3[tid] + b3v;
      float move  = ego_v * 0.1f + 0.5f * a_new * 0.01f;   // DT=0.1
      disp  += move;
      ego_x += move;
      ego_v += a_new * 0.1f;
      ego_a  = a_new;
      outAct [(size_t)b * T_STEPS + t]           = a_new;
      outDisp[(size_t)b * (T_STEPS + 1) + t + 1] = disp;
    }
    // no 4th barrier: (C) + (A of t+1) already separate every read(t) from write(t+1)
  }
}

extern "C" void kernel_launch(void* const* d_in, const int* in_sizes, int n_in,
                              void* d_out, int out_size, void* d_ws, size_t ws_size,
                              hipStream_t stream) {
  const float* proj = (const float*)d_in[0];
  const float* enc  = (const float*)d_in[1];
  const float* idm  = (const float*)d_in[2];
  const float* mcg  = (const float*)d_in[3];
  const float* em   = (const float*)d_in[4];
  const float* ev   = (const float*)d_in[5];
  const float* W1   = (const float*)d_in[6];
  const float* b1   = (const float*)d_in[7];
  const float* W2   = (const float*)d_in[8];
  const float* b2   = (const float*)d_in[9];
  const float* W3   = (const float*)d_in[10];
  const float* b3   = (const float*)d_in[11];

  float* outDisp = (float*)d_out;                                   // B*(T+1)
  float* outAct  = outDisp + (size_t)B_TOT * (T_STEPS + 1);         // then B*T

  dim3 grid(B_TOT / NROWS);   // 512 workgroups
  dim3 block(NTHREADS);       // 8 waves
  forward_sim_kernel<<<grid, block, 0, stream>>>(proj, enc, idm, mcg, em, ev,
                                                 W1, b1, W2, b2, W3, b3,
                                                 outDisp, outAct);
}
